// AtencaoCruzada_74887049773587
// MI455X (gfx1250) — matrix operand order repaired and checked by reading, not compiled, not run on hardware
//
#include <hip/hip_runtime.h>
#include <cstdint>
#include <cstddef>

// ---------------------------------------------------------------------------
// CDNA5 (gfx1250) wave32 WMMA types
// ---------------------------------------------------------------------------
typedef _Float16 v16h __attribute__((ext_vector_type(16)));
typedef float    v8f  __attribute__((ext_vector_type(8)));
typedef unsigned int v4u __attribute__((ext_vector_type(4)));
typedef int          v4i __attribute__((ext_vector_type(4)));
typedef int          v8i __attribute__((ext_vector_type(8)));

// Trivial POD 16B/8B carriers (HIP's uint4/float4 have ctors -> unusable in unions)
struct alignas(16) U4 { unsigned int x, y, z, w; };
struct alignas(16) F4 { float x, y, z, w; };
struct alignas(8)  U2 { unsigned int x, y; };

union FragU { v16h v; U4 u[2]; };
union H4    { _Float16 h[4]; U2 u; };

// Tensor Data Mover availability (device pass only; arity differs per toolchain)
#if defined(__AMDGCN__) && __has_builtin(__builtin_amdgcn_tensor_load_to_lds) && \
    __has_builtin(__builtin_amdgcn_s_wait_tensorcnt)
#define USE_TDM 1
#else
#define USE_TDM 0
#endif

// A-matrix 16x32 f16 fragment: per lane, elements 0..15 = K=kb..kb+15
// (kb = 0 for lanes 0-15, 16 for lanes 16-31).
__device__ __forceinline__ v16h ld_frag_a16(const _Float16* p) {
  FragU f;
  f.u[0] = *(const U4*)(p);
  f.u[1] = *(const U4*)(p + 8);
  return f.v;
}
// B-matrix 32x16 f16 fragment: lane = column n, elements 0..15 = K=klo..klo+15
// (klo = 0 for lanes 0-15, 16 for lanes 16-31) -> 32 contiguous bytes of B^T row.
__device__ __forceinline__ v16h ld_frag_b16(const _Float16* p) {
  FragU f;
  f.u[0] = *(const U4*)(p);
  f.u[1] = *(const U4*)(p + 8);
  return f.v;
}

#define WMMA_F16(A, B, C) \
  __builtin_amdgcn_wmma_f32_16x16x32_f16(false, (A), false, (B), (short)0, (C), false, false)

#if USE_TDM
// Issue a 2D Tensor-Data-Mover load: tile_h rows of tile_w f16 elements
// (row stride = stride_elems), global -> LDS, with hardware row padding
// (pad_amount_code+1 DWORDs appended every 2^(pad_interval_code+1) DWORDs).
// D# layout per CDNA5 ISA.
__device__ __forceinline__ void tdm_load_2d(
    unsigned lds_off, const void* gptr,
    unsigned tile_w, unsigned tile_h, unsigned stride_elems,
    unsigned pad_interval_code, unsigned pad_amount_code)
{
  unsigned long long ga = (unsigned long long)(uintptr_t)gptr;
  v4u g0;
  g0[0] = 1u;                                               // count=1, user desc
  g0[1] = lds_off;                                          // lds_addr (bytes)
  g0[2] = (unsigned)ga;                                     // global_addr[31:0]
  g0[3] = (unsigned)((ga >> 32) & 0x1FFFFFFu) | (2u << 30); // addr[56:32] | type=2
  v8i g1;
  g1[0] = (int)((1u << 16) | (1u << 20) |                   // data_size=2B, pad_enable
                (pad_interval_code << 22) | (pad_amount_code << 25));
  g1[1] = (int)(tile_w << 16);        // tensor_dim0[15:0] << 16
  g1[2] = (int)(tile_h << 16);        // tensor_dim0 hi=0 | tensor_dim1[15:0] << 16
  g1[3] = (int)(tile_w << 16);        // tensor_dim1 hi=0 | tile_dim0
  g1[4] = (int)tile_h;                // tile_dim1 (tile_dim2=0)
  g1[5] = (int)stride_elems;          // tensor_dim0_stride[31:0]
  g1[6] = 0;
  g1[7] = 0;
  v4i gz = {0, 0, 0, 0};
#if __clang_major__ >= 23
  v8i gz8 = {0, 0, 0, 0, 0, 0, 0, 0};
  __builtin_amdgcn_tensor_load_to_lds(g0, g1, gz, gz, gz8, 0);
#else
  __builtin_amdgcn_tensor_load_to_lds(g0, g1, gz, gz, 0);
#endif
}
#endif

// ---------------------------------------------------------------------------
// GEMM:  C[M,N] = A[M,K] @ W[K,N] + bias[N]
//   A: f32 (Af) or f16 (Ah).   C: f32 (cmode 0), f16 (cmode 1),
//   f16 transposed-per-batch (cmode 2: C[b][n][r], r = row within batch).
// Block: 256 threads = 8 waves; block tile 128x64; wave tile 16x64; K-step 32.
// Software-pipelined: next K-slab is loaded to registers while WMMA consumes
// the current slab from LDS, so s_wait_loadcnt lands behind the matrix work.
// ---------------------------------------------------------------------------
#define BM 128
#define BN 64
#define BK 32
#define APITCH 40   // 32 halves + 8 pad = 80B row pitch (16B aligned)
#define BPITCH 40

__global__ __launch_bounds__(256) void gemm_bias_wmma(
    const float* __restrict__ Af, const _Float16* __restrict__ Ah,
    const float* __restrict__ W, const float* __restrict__ bias,
    float* __restrict__ Cf, _Float16* __restrict__ Ch,
    int cmode, int M, int K, int N, int rows_per_batch)
{
  __shared__ alignas(16) _Float16 As[BM * APITCH];   // 10.0 KB
  __shared__ alignas(16) _Float16 Bt[BN * BPITCH];   //  5.0 KB (W tile, transposed)

  const int tid   = threadIdx.x;
  const int lane  = tid & 31;
  const int wave  = tid >> 5;
  const int m0blk = blockIdx.y * BM;
  const int n0blk = blockIdx.x * BN;
  const int half  = lane >> 4;
  const int l16   = lane & 15;

  const v8f z8 = {0.f, 0.f, 0.f, 0.f, 0.f, 0.f, 0.f, 0.f};
  v8f acc[4];
#pragma unroll
  for (int t = 0; t < 4; ++t) acc[t] = z8;

  // register staging for the in-flight K-slab
  F4 aF[4];
  U4 aH[2];
  F4 wF[2];

  auto load_tiles = [&](int k0) {
    if (Af) {
#pragma unroll
      for (int e = 0; e < 4; ++e) {
        int idx = tid + e * 256;                 // 1024 float4 slots over 128x32
        int row = idx >> 3;
        int seg = (idx & 7) << 2;
        aF[e] = *(const F4*)(Af + (size_t)(m0blk + row) * K + k0 + seg);
      }
    } else {
#pragma unroll
      for (int e = 0; e < 2; ++e) {
        int idx = tid + e * 256;                 // 512 8-half slots
        int row = idx >> 2;
        int seg = (idx & 3) << 3;
        aH[e] = *(const U4*)(Ah + (size_t)(m0blk + row) * K + k0 + seg);
      }
    }
#pragma unroll
    for (int e = 0; e < 2; ++e) {
      int idx = tid + e * 256;                   // 512 float4 slots over 32x64
      int row = idx >> 4;
      int c4  = (idx & 15) << 2;
      wF[e] = *(const F4*)(W + (size_t)(k0 + row) * N + n0blk + c4);
    }
  };

  auto store_tiles = [&]() {
    if (Af) {
#pragma unroll
      for (int e = 0; e < 4; ++e) {
        int idx = tid + e * 256;
        int row = idx >> 3;
        int seg = (idx & 7) << 2;
        H4 p;
        p.h[0] = (_Float16)aF[e].x; p.h[1] = (_Float16)aF[e].y;
        p.h[2] = (_Float16)aF[e].z; p.h[3] = (_Float16)aF[e].w;
        *(U2*)&As[row * APITCH + seg] = p.u;
      }
    } else {
#pragma unroll
      for (int e = 0; e < 2; ++e) {
        int idx = tid + e * 256;
        int row = idx >> 2;
        int seg = (idx & 3) << 3;
        *(U4*)&As[row * APITCH + seg] = aH[e];
      }
    }
#pragma unroll
    for (int e = 0; e < 2; ++e) {
      int idx = tid + e * 256;
      int row = idx >> 4;
      int c4  = (idx & 15) << 2;
      Bt[(c4 + 0) * BPITCH + row] = (_Float16)wF[e].x;
      Bt[(c4 + 1) * BPITCH + row] = (_Float16)wF[e].y;
      Bt[(c4 + 2) * BPITCH + row] = (_Float16)wF[e].z;
      Bt[(c4 + 3) * BPITCH + row] = (_Float16)wF[e].w;
    }
  };

  load_tiles(0);
  for (int k0 = 0; k0 < K; k0 += BK) {
    store_tiles();
    __syncthreads();
    if (k0 + BK < K) {
      load_tiles(k0 + BK);                       // overlaps with WMMA below
      if (k0 + 2 * BK < K) {                     // -> global_prefetch_b8
        if (Af) __builtin_prefetch(Af + (size_t)(m0blk + (tid >> 1)) * K + k0 + 2 * BK, 0, 0);
        else    __builtin_prefetch(Ah + (size_t)(m0blk + (tid >> 1)) * K + k0 + 2 * BK, 0, 0);
        __builtin_prefetch(W + (size_t)(k0 + 2 * BK + (tid >> 3)) * N + n0blk, 0, 0);
      }
    }

    // ---- wave computes its 16x64 strip: 1 A-frag, 4 wmma ----
    v16h afrag = ld_frag_a16(&As[(wave * 16 + l16) * APITCH + half * 16]);
#pragma unroll
    for (int t = 0; t < 4; ++t) {
      v16h bfrag = ld_frag_b16(&Bt[(t * 16 + l16) * BPITCH + half * 16]);
      acc[t] = WMMA_F16(afrag, bfrag, acc[t]);
    }
    __syncthreads();
  }

  // ---- epilogue: bias + store (C-fragment: lane holds rows r+half*8, col l16) ----
  const int mrow0 = m0blk + wave * 16 + half * 8;
#pragma unroll
  for (int t = 0; t < 4; ++t) {
    const int n = n0blk + t * 16 + l16;
    const float bv = bias[n];
#pragma unroll
    for (int r = 0; r < 8; ++r) {
      const int m = mrow0 + r;
      const float val = acc[t][r] + bv;
      if (cmode == 0) {
        Cf[(size_t)m * N + n] = val;
      } else if (cmode == 1) {
        Ch[(size_t)m * N + n] = (_Float16)val;
      } else {                                   // per-batch transposed f16 (V^T)
        const int bidx = m / rows_per_batch;
        const int rr   = m - bidx * rows_per_batch;
        Ch[((size_t)bidx * N + n) * rows_per_batch + rr] = (_Float16)val;
      }
    }
  }
}

// ---------------------------------------------------------------------------
// Flash-style cross attention.  D_EMB=1024, D_HEAD=64 fixed.
// Grid: (Sq/64, B*H); block = 128 threads = 4 waves; each wave: 16 q-rows.
// K and V chunks are staged global->LDS by the Tensor Data Mover (2D D#
// descriptors with hardware row padding that reproduces the bank-spreading
// LDS pitches), issued by waves 0/1 in parallel, fenced with s_wait_tensorcnt
// + workgroup barrier.  Inner loop issues no VMEM at all.
// ---------------------------------------------------------------------------
#define KPITCH 72   // 64 halves + 8 pad (TDM: pad 4dw per 32dw row)
#define VPITCH 40   // 32 halves + 8 pad (TDM: pad 4dw per 16dw row)
#define PPITCH 40

__global__ __launch_bounds__(128) void attn_wmma(
    const _Float16* __restrict__ Q, const _Float16* __restrict__ Kh,
    const _Float16* __restrict__ Vt, _Float16* __restrict__ O,
    int Sq, int Sk)
{
  __shared__ alignas(16) _Float16 Ks[32 * KPITCH];       // K chunk  [key][d]
  __shared__ alignas(16) _Float16 Vs[64 * VPITCH];       // V chunk  [d][key]  (=B^T)
  __shared__ alignas(16) _Float16 Ps[4 * 16 * PPITCH];   // per-wave P staging

  const int tid  = threadIdx.x;
  const int lane = tid & 31;
  const int wave = tid >> 5;
  const int half = lane >> 4;
  const int l16  = lane & 15;
  const int b = blockIdx.y >> 4;
  const int h = blockIdx.y & 15;
  const int qrow0 = b * Sq + blockIdx.x * 64 + wave * 16;

  // Q fragments for this wave's 16 rows (2 d-chunks of 32), loaded once.
  v16h qf[2];
  {
    const _Float16* qbase = Q + (size_t)(qrow0 + l16) * 1024 + h * 64;
#pragma unroll
    for (int c = 0; c < 2; ++c)
      qf[c] = ld_frag_a16(qbase + c * 32 + half * 16);
  }

  const v8f z8 = {0.f, 0.f, 0.f, 0.f, 0.f, 0.f, 0.f, 0.f};
  v8f o[4];
#pragma unroll
  for (int t = 0; t < 4; ++t) o[t] = z8;
  float mrow[8], lrow[8];
#pragma unroll
  for (int r = 0; r < 8; ++r) { mrow[r] = -3.0e38f; lrow[r] = 0.f; }

#if USE_TDM
  const unsigned ldsKs = (unsigned)(uintptr_t)(&Ks[0]);  // LDS byte offset (addr[31:0])
  const unsigned ldsVs = (unsigned)(uintptr_t)(&Vs[0]);
#endif

  for (int k0 = 0; k0 < Sk; k0 += 32) {
#if USE_TDM
    // ---- TDM: K chunk 32(keys) x 64(d), V^T chunk 64(d) x 32(keys) ----
    if (wave == 0) {
      tdm_load_2d(ldsKs, Kh + (size_t)(b * Sk + k0) * 1024 + h * 64,
                  /*tile_w=*/64, /*tile_h=*/32, /*stride=*/1024,
                  /*pad_interval: 32dw*/4, /*pad: 4dw*/3);
    } else if (wave == 1) {
      tdm_load_2d(ldsVs, Vt + ((size_t)b * 1024 + h * 64) * Sk + k0,
                  /*tile_w=*/32, /*tile_h=*/64, /*stride=*/(unsigned)Sk,
                  /*pad_interval: 16dw*/3, /*pad: 4dw*/3);
    }
    if (wave < 2) __builtin_amdgcn_s_wait_tensorcnt((short)0);
#else
    // ---- fallback: cooperative vector loads ----
    {
      const int key  = tid >> 2;
      const int dseg = (tid & 3) << 4;
      const _Float16* src = Kh + (size_t)(b * Sk + k0 + key) * 1024 + h * 64 + dseg;
      U4 ka = *(const U4*)(src);
      U4 kb2 = *(const U4*)(src + 8);
      *(U4*)&Ks[key * KPITCH + dseg]     = ka;
      *(U4*)&Ks[key * KPITCH + dseg + 8] = kb2;
    }
    {
      const int d    = tid >> 1;
      const int kseg = (tid & 1) << 4;
      const _Float16* src = Vt + ((size_t)b * 1024 + h * 64 + d) * Sk + k0 + kseg;
      U4 va = *(const U4*)(src);
      U4 vb = *(const U4*)(src + 8);
      *(U4*)&Vs[d * VPITCH + kseg]     = va;
      *(U4*)&Vs[d * VPITCH + kseg + 8] = vb;
    }
    if (k0 + 32 < Sk) {
      __builtin_prefetch(Vt + ((size_t)b * 1024 + h * 64 + (tid >> 1)) * Sk + k0 + 32, 0, 0);
    }
#endif
    __syncthreads();

    // ---- scores: two 16(q) x 16(key) tiles, accumulated over d (2 wmma each) ----
    v8f s[2];
#pragma unroll
    for (int jt = 0; jt < 2; ++jt) {
      v8f sc = z8;
#pragma unroll
      for (int c = 0; c < 2; ++c) {
        v16h bf = ld_frag_b16(&Ks[(jt * 16 + l16) * KPITCH + c * 32 + half * 16]);
        sc = WMMA_F16(qf[c], bf, sc);
      }
      s[jt] = sc;
    }

    // ---- online softmax (row r lives at C-fragment element r, half-dependent) ----
#pragma unroll
    for (int r = 0; r < 8; ++r) {
      float v0 = s[0][r] * 0.125f;      // 1/sqrt(64)
      float v1 = s[1][r] * 0.125f;
      float cm = fmaxf(v0, v1);
#pragma unroll
      for (int msk = 1; msk < 16; msk <<= 1) cm = fmaxf(cm, __shfl_xor(cm, msk, 32));
      const float mnew  = fmaxf(mrow[r], cm);
      const float alpha = __expf(mrow[r] - mnew);
      const float p0 = __expf(v0 - mnew);
      const float p1 = __expf(v1 - mnew);
      float psum = p0 + p1;
#pragma unroll
      for (int msk = 1; msk < 16; msk <<= 1) psum += __shfl_xor(psum, msk, 32);
      lrow[r] = lrow[r] * alpha + psum;
      mrow[r] = mnew;
#pragma unroll
      for (int t = 0; t < 4; ++t) o[t][r] *= alpha;
      s[0][r] = p0;
      s[1][r] = p1;
    }

    // ---- P: C-layout -> A-layout via same-wave LDS round trip (DS in-order) ----
    {
      _Float16* pw = &Ps[wave * 16 * PPITCH];
#pragma unroll
      for (int jt = 0; jt < 2; ++jt)
#pragma unroll
        for (int r = 0; r < 8; ++r)
          pw[(half * 8 + r) * PPITCH + jt * 16 + l16] = (_Float16)s[jt][r];
    }
    {
      v16h pf = ld_frag_a16(&Ps[(wave * 16 + l16) * PPITCH + half * 16]);
#pragma unroll
      for (int t = 0; t < 4; ++t) {
        v16h vf = ld_frag_b16(&Vs[(t * 16 + l16) * VPITCH + half * 16]);
        o[t] = WMMA_F16(pf, vf, o[t]);
      }
    }
    __syncthreads();
  }

  // ---- normalize and store attn output [row][h*64+d] as f16 ----
#pragma unroll
  for (int t = 0; t < 4; ++t) {
#pragma unroll
    for (int r = 0; r < 8; ++r) {
      const int m = qrow0 + half * 8 + r;
      const int n = h * 64 + t * 16 + l16;
      O[(size_t)m * 1024 + n] = (_Float16)(o[t][r] / lrow[r]);
    }
  }
}

// ---------------------------------------------------------------------------
// Host launch
// ---------------------------------------------------------------------------
extern "C" void kernel_launch(void* const* d_in, const int* in_sizes, int n_in,
                              void* d_out, int out_size, void* d_ws, size_t ws_size,
                              hipStream_t stream) {
  (void)in_sizes; (void)n_in; (void)out_size; (void)ws_size;

  const float* x  = (const float*)d_in[0];
  const float* y  = (const float*)d_in[1];
  const float* Wq = (const float*)d_in[2];
  const float* bq = (const float*)d_in[3];
  const float* Wk = (const float*)d_in[4];
  const float* bk = (const float*)d_in[5];
  const float* Wv = (const float*)d_in[6];
  const float* bv = (const float*)d_in[7];
  const float* Wo = (const float*)d_in[8];
  const float* bo = (const float*)d_in[9];
  float* out = (float*)d_out;

  const int B = 4, Sq = 4096, Sk = 1024, D = 1024, DC = 768, H = 16;
  const int Mq = B * Sq;   // 16384
  const int Mk = B * Sk;   // 4096

  char* ws = (char*)d_ws;
  _Float16* qh  = (_Float16*)ws;  ws += (size_t)Mq * D * sizeof(_Float16);     // 32 MB
  _Float16* kh  = (_Float16*)ws;  ws += (size_t)Mk * D * sizeof(_Float16);     //  8 MB
  _Float16* vth = (_Float16*)ws;  ws += (size_t)B * D * Sk * sizeof(_Float16); //  8 MB
  _Float16* ah  = (_Float16*)ws;  // 32 MB

  dim3 blk(256);
  // Q = x@Wq + bq -> f16
  gemm_bias_wmma<<<dim3(D / BN, Mq / BM), blk, 0, stream>>>(
      x, nullptr, Wq, bq, nullptr, qh, 1, Mq, D, D, 0);
  // K = y@Wk + bk -> f16
  gemm_bias_wmma<<<dim3(D / BN, Mk / BM), blk, 0, stream>>>(
      y, nullptr, Wk, bk, nullptr, kh, 1, Mk, DC, D, 0);
  // V = y@Wv + bv -> f16, stored transposed per batch: Vt[b][n][key]
  gemm_bias_wmma<<<dim3(D / BN, Mk / BM), blk, 0, stream>>>(
      y, nullptr, Wv, bv, nullptr, vth, 2, Mk, DC, D, Sk);
  // flash attention -> f16 [Mq][1024]
  attn_wmma<<<dim3(Sq / 64, B * H), dim3(128), 0, stream>>>(qh, kh, vth, ah, Sq, Sk);
  // out = attn@Wo + bo -> f32
  gemm_bias_wmma<<<dim3(D / BN, Mq / BM), blk, 0, stream>>>(
      nullptr, ah, Wo, bo, out, nullptr, 0, Mq, D, D, 0);
}
